// MultiHeadSelfMSR_69037304316601
// MI455X (gfx1250) — compile-verified, matrix operand order checked by reading, not run on hardware
//
#include <hip/hip_runtime.h>
#include <cstdint>

typedef __attribute__((ext_vector_type(16))) _Float16 v16h;
typedef __attribute__((ext_vector_type(8)))  float    v8f;

#define LDP 72                      // LDS row pitch in halves (64+8; 144B keeps 16B alignment)
#define LOG2_GAMMA (-0.15200309344504995f)   // log2(0.9)

__device__ __forceinline__ v8f wmma_f16(v16h a, v16h b, v8f c) {
  return __builtin_amdgcn_wmma_f32_16x16x32_f16(false, a, false, b, (short)0, c, false, false);
}

// CDNA5 16-bit A/B layout: lanes 0-15 hold K = {0..7,16..23}, lanes 16-31 K = {8..15,24..31}
__device__ __forceinline__ int kpair_base(int v, int hi) {
  return 2 * v + (v >= 4 ? 8 : 0) + 8 * hi;
}

// Fragment: lane walks rows, K walks contiguous columns -> 2x ds_load_b128.
__device__ __forceinline__ v16h frag_row(const _Float16* sm, int row0, int col0, int lane) {
  int m = lane & 15, hi = lane >> 4;
  v16h f;
#pragma unroll
  for (int v = 0; v < 8; ++v) {
    int kb = kpair_base(v, hi);
    const _Float16* p = sm + (row0 + m) * LDP + col0 + kb;
    f[2 * v]     = p[0];
    f[2 * v + 1] = p[1];
  }
  return f;
}

// Synchronous 64x64 f16 tile copy (transpose kernels only).
__device__ __forceinline__ void load_tile64(_Float16* dst, const _Float16* src, int src_ld, int tid) {
#pragma unroll
  for (int it = 0; it < 8; ++it) {
    int idx = tid + it * 256;
    int r  = idx >> 5;
    int c2 = (idx & 31) << 1;
    *(uint32_t*)(dst + r * LDP + c2) =
        *(const uint32_t*)(src + (size_t)r * src_ld + c2);
  }
}

// ---- CDNA5 async global->LDS (ASYNCcnt), GVS addressing: saddr base + 32-bit voffset ----
__device__ __forceinline__ void async_b128(uint32_t lds_byte_off, uint32_t voff, const void* sbase) {
  asm volatile("global_load_async_to_lds_b128 %0, %1, %2"
               :: "v"(lds_byte_off), "v"(voff), "s"(sbase) : "memory");
}
__device__ __forceinline__ uint32_t lds_addr32(const void* p) {
  return (uint32_t)(size_t)p;   // generic LDS pointer: low 32 bits = wave-relative LDS offset
}
#define WAIT_ASYNC_4() asm volatile("s_wait_asynccnt 4" ::: "memory")
#define WAIT_ASYNC_0() asm volatile("s_wait_asynccnt 0" ::: "memory")

// Issue one 64x64 f16 tile (8 KB) as 2 async b128 per thread; rows stride row_stride_bytes.
__device__ __forceinline__ void issue_tile_async(_Float16* dst, const _Float16* base,
                                                 int row_stride_bytes, int tid) {
  uint32_t lds = lds_addr32(dst);
#pragma unroll
  for (int j = 0; j < 2; ++j) {
    int idx = tid + j * 256;            // 0..511 b128 chunks
    int r   = idx >> 3;
    int c16 = (idx & 7) * 16;
    async_b128(lds + r * (LDP * 2) + c16, (uint32_t)(r * row_stride_bytes + c16), base);
  }
}

__global__ void __launch_bounds__(256) f32_to_f16_kernel(const float* __restrict__ in,
                                                         _Float16* __restrict__ out, int n) {
  int i = blockIdx.x * 256 + threadIdx.x;
  if (i < n) out[i] = (_Float16)in[i];
}

// -------- Weight convert + transpose: WhT[z][n][k] (f16) from W[z][k][n] (f32) ------------
__global__ void __launch_bounds__(256) convert_transpose_w(const float* __restrict__ Wq,
                                                           const float* __restrict__ Wk,
                                                           const float* __restrict__ Wv,
                                                           _Float16* __restrict__ WhT) {
  __shared__ _Float16 t[64 * LDP];
  int z = blockIdx.z;
  const float* W = (z == 0) ? Wq : ((z == 1) ? Wk : Wv);
  int n0 = blockIdx.x * 64, k0 = blockIdx.y * 64;
  int tid = threadIdx.x;
#pragma unroll
  for (int it = 0; it < 16; ++it) {
    int idx = tid + it * 256;
    int r = idx >> 6, c = idx & 63;
    t[c * LDP + r] = (_Float16)W[(size_t)(k0 + r) * 512 + n0 + c];
  }
  __syncthreads();
  _Float16* out = WhT + (size_t)z * 512 * 512;
#pragma unroll
  for (int it = 0; it < 8; ++it) {
    int idx = tid + it * 256;
    int r = idx >> 5, c2 = (idx & 31) << 1;
    *(uint32_t*)(out + (size_t)(n0 + r) * 512 + k0 + c2) = *(uint32_t*)(t + r * LDP + c2);
  }
}

// -------- V head-tile transpose: vT[b,h][d][s] from vh[b,h][s][d] ------------------------
__global__ void __launch_bounds__(256) transpose_v(const _Float16* __restrict__ vh,
                                                   _Float16* __restrict__ vT, int S) {
  __shared__ _Float16 t[64 * LDP];
  int s0 = blockIdx.x * 64;
  size_t hb = (size_t)blockIdx.y * (size_t)S * 64;
  int tid = threadIdx.x;
#pragma unroll
  for (int it = 0; it < 8; ++it) {
    int idx = tid + it * 256;
    int r = idx >> 5, c2 = (idx & 31) << 1;
    uint32_t pr = *(const uint32_t*)(vh + hb + (size_t)(s0 + r) * 64 + c2);
    union { uint32_t u; _Float16 h[2]; } cv; cv.u = pr;
    t[(c2    ) * LDP + r] = cv.h[0];
    t[(c2 + 1) * LDP + r] = cv.h[1];
  }
  __syncthreads();
  _Float16* out = vT + hb;
#pragma unroll
  for (int it = 0; it < 8; ++it) {
    int idx = tid + it * 256;
    int r = idx >> 5, c2 = (idx & 31) << 1;
    *(uint32_t*)(out + (size_t)r * S + s0 + c2) = *(uint32_t*)(t + r * LDP + c2);
  }
}

// -------- Projection GEMM (async double-buffered): out[z] = Xh @ W[z], f32 out -----------
__global__ void __launch_bounds__(256) proj_gemm_f16(const _Float16* __restrict__ Xh,
                                                     const _Float16* __restrict__ WhT,
                                                     float* __restrict__ outf,
                                                     int M, int N, int K) {
  __shared__ _Float16 xa[2][64 * LDP];
  __shared__ _Float16 wt[2][64 * LDP];
  int n0 = blockIdx.x * 64, m0 = blockIdx.y * 64, z = blockIdx.z;
  const _Float16* WT = WhT + (size_t)z * K * N;
  float* out = outf + (size_t)z * M * N;
  int tid = threadIdx.x, lane = tid & 31, w = tid >> 5;
  int mi = w >> 1;
  int nb = (w & 1) * 2;
  v8f acc0 = {}; v8f acc1 = {};

  issue_tile_async(xa[0], Xh + (size_t)m0 * K, K * 2, tid);
  issue_tile_async(wt[0], WT + (size_t)n0 * K, K * 2, tid);
  const int NIT = 8;                        // K / 64
  for (int i = 0; i < NIT; ++i) {
    if (i + 1 < NIT) {
      int kk = (i + 1) * 64;
      issue_tile_async(xa[(i + 1) & 1], Xh + (size_t)m0 * K + kk, K * 2, tid);
      issue_tile_async(wt[(i + 1) & 1], WT + (size_t)n0 * K + kk, K * 2, tid);
      WAIT_ASYNC_4();                       // drain step i (in-order), keep step i+1 in flight
    } else {
      WAIT_ASYNC_0();
    }
    __syncthreads();                        // step i visible across waves
    const _Float16* xc = xa[i & 1];
    const _Float16* wc = wt[i & 1];
    v16h a0  = frag_row(xc, mi * 16, 0,  lane);
    v16h a1  = frag_row(xc, mi * 16, 32, lane);
    v16h b00 = frag_row(wc, nb * 16, 0,  lane);
    v16h b01 = frag_row(wc, nb * 16, 32, lane);
    v16h b10 = frag_row(wc, (nb + 1) * 16, 0,  lane);
    v16h b11 = frag_row(wc, (nb + 1) * 16, 32, lane);
    acc0 = wmma_f16(a0, b00, acc0);
    acc0 = wmma_f16(a1, b01, acc0);
    acc1 = wmma_f16(a0, b10, acc1);
    acc1 = wmma_f16(a1, b11, acc1);
    __syncthreads();                        // reads done before buffer is re-streamed
  }
  int n = lane & 15, hi = lane >> 4;
  float* op = out + (size_t)(m0 + mi * 16 + 8 * hi) * N + n0 + nb * 16 + n;
#pragma unroll
  for (int r = 0; r < 8; ++r) {
    op[0]  = acc0[r];
    op[16] = acc1[r];
    op += N;
  }
}

// -------- xpos + GroupNorm(k) + SiLU(q), write head-major f16 ----------------------------
__global__ void __launch_bounds__(512) xpos_gn_silu(const float* __restrict__ Qf,
                                                    const float* __restrict__ Kf,
                                                    const float* __restrict__ Vf,
                                                    const float* __restrict__ gnw,
                                                    const float* __restrict__ gnb,
                                                    _Float16* __restrict__ qh,
                                                    _Float16* __restrict__ kh,
                                                    _Float16* __restrict__ vh, int S) {
  const int Dm = 512;
  int s = blockIdx.x, b = blockIdx.y, c = threadIdx.x;
  size_t rowb = ((size_t)b * S + s) * Dm;
  float qv = Qf[rowb + c], kv = Kf[rowb + c], vv = Vf[rowb + c];
  float qp = Qf[rowb + (c ^ 1)], kp = Kf[rowb + (c ^ 1)];
  int i = c >> 1;
  float sc   = powf((2.0f * i + 0.4f * Dm) / (1.4f * Dm), (float)s / 512.0f);
  float invf = exp2f(-(float)i * (13.28771237954945f / 256.0f));
  float ang  = (float)s * invf;
  float sn = sinf(ang) * sc, cs = cosf(ang) * sc;
  float sgn = (c & 1) ? 1.0f : -1.0f;
  float qx = qv * cs + sgn * qp * sn;
  float kx = kv * cs + sgn * kp * sn;
  float s1 = kx, s2 = kx * kx;
#pragma unroll
  for (int off = 16; off; off >>= 1) {
    s1 += __shfl_xor(s1, off, 32);
    s2 += __shfl_xor(s2, off, 32);
  }
  __shared__ float red[32];
  int wid = c >> 5;
  if ((c & 31) == 0) { red[wid * 2] = s1; red[wid * 2 + 1] = s2; }
  __syncthreads();
  s1 += red[(wid ^ 1) * 2];
  s2 += red[(wid ^ 1) * 2 + 1];
  float mu  = s1 * (1.0f / 64.0f);
  float var = s2 * (1.0f / 64.0f) - mu * mu;
  float kn = (kx - mu) * rsqrtf(var + 1e-5f) * gnw[c] + gnb[c];
  float qsilu = qx / (1.0f + expf(-qx));
  int h = c >> 6, d = c & 63;
  size_t o = (((size_t)b * 8 + h) * S + s) * 64 + d;
  qh[o] = (_Float16)qsilu;
  kh[o] = (_Float16)kn;
  vh[o] = (_Float16)vv;
}

// -------- Fused retention: fully async loads, double-buffered K/V/P, 2 barriers/chunk -----
__global__ void __launch_bounds__(256) attn_msr(const _Float16* __restrict__ qh,
                                                const _Float16* __restrict__ kh,
                                                const _Float16* __restrict__ vT,
                                                const float* __restrict__ X,
                                                float* __restrict__ out, int S) {
  __shared__ _Float16 qs[64 * LDP];
  __shared__ _Float16 ps[2][64 * LDP];
  __shared__ _Float16 ksb[2][64 * LDP];
  __shared__ _Float16 vsb[2][64 * LDP];      // d-major: vs[d][t]
  int rb = blockIdx.x, h = blockIdx.y, b = blockIdx.z;
  int tid = threadIdx.x, lane = tid & 31, w = tid >> 5;
  int mi = w >> 1;
  int nb = (w & 1) * 2;
  int n = lane & 15, hi = lane >> 4;
  size_t hb = ((size_t)(b * 8 + h)) * (size_t)S * 64;

  int tc0 = rb - 5; if (tc0 < 0) tc0 = 0;    // gamma^321 < 2e-15: 6 chunks suffice (fp32-exact)

  // prologue: qs (2 asyncs) then chunk tc0 (4 asyncs); wait<=4 drains exactly qs
  issue_tile_async(qs, qh + hb + (size_t)rb * 64 * 64, 128, tid);
  issue_tile_async(ksb[tc0 & 1], kh + hb + (size_t)tc0 * 64 * 64, 128, tid);
  issue_tile_async(vsb[tc0 & 1], vT + hb + (size_t)tc0 * 64, S * 2, tid);
  WAIT_ASYNC_4();
  __syncthreads();                            // qs visible to all waves
  v16h aq0 = frag_row(qs, mi * 16, 0,  lane); // loop-invariant Q fragments
  v16h aq1 = frag_row(qs, mi * 16, 32, lane);

  v8f o0 = {}; v8f o1 = {};
  for (int tc = tc0; tc <= rb; ++tc) {
    const _Float16* kcur = ksb[tc & 1];
    const _Float16* vcur = vsb[tc & 1];
    _Float16* pcur = ps[tc & 1];
    if (tc < rb) {
      issue_tile_async(ksb[(tc + 1) & 1], kh + hb + (size_t)(tc + 1) * 64 * 64, 128, tid);
      issue_tile_async(vsb[(tc + 1) & 1], vT + hb + (size_t)(tc + 1) * 64, S * 2, tid);
      WAIT_ASYNC_4();                         // drain chunk tc, keep tc+1 in flight
    } else {
      WAIT_ASYNC_0();
    }
    __syncthreads();                          // chunk tc visible across all waves

    // all K and V fragment reads happen before the mid barrier (buffer-reuse fence)
    v16h b00 = frag_row(kcur, nb * 16, 0,  lane);
    v16h b01 = frag_row(kcur, nb * 16, 32, lane);
    v16h b10 = frag_row(kcur, (nb + 1) * 16, 0,  lane);
    v16h b11 = frag_row(kcur, (nb + 1) * 16, 32, lane);
    v16h v00 = frag_row(vcur, nb * 16, 0,  lane);
    v16h v01 = frag_row(vcur, nb * 16, 32, lane);
    v16h v10 = frag_row(vcur, (nb + 1) * 16, 0,  lane);
    v16h v11 = frag_row(vcur, (nb + 1) * 16, 32, lane);
    v8f s0 = {}; v8f s1 = {};
    s0 = wmma_f16(aq0, b00, s0);
    s0 = wmma_f16(aq1, b01, s0);
    s1 = wmma_f16(aq0, b10, s1);
    s1 = wmma_f16(aq1, b11, s1);

    // causal gamma^(s-t) decay in f32, stash as f16 into this chunk's P buffer
#pragma unroll
    for (int r = 0; r < 8; ++r) {
      int m = r + 8 * hi;
      int srow = rb * 64 + mi * 16 + m;
      int t0 = tc * 64 + nb * 16 + n;
      int t1 = t0 + 16;
      float d0 = (srow >= t0) ? exp2f((float)(srow - t0) * LOG2_GAMMA) : 0.0f;
      float d1 = (srow >= t1) ? exp2f((float)(srow - t1) * LOG2_GAMMA) : 0.0f;
      pcur[(mi * 16 + m) * LDP + nb * 16 + n]       = (_Float16)(s0[r] * d0);
      pcur[(mi * 16 + m) * LDP + (nb + 1) * 16 + n] = (_Float16)(s1[r] * d1);
    }
    __syncthreads();                          // P visible; also fences K/V buffer reuse

    // O += P(64x64) @ V(64xhd)
    v16h p0 = frag_row(pcur, mi * 16, 0,  lane);
    v16h p1 = frag_row(pcur, mi * 16, 32, lane);
    o0 = wmma_f16(p0, v00, o0);
    o0 = wmma_f16(p1, v01, o0);
    o1 = wmma_f16(p0, v10, o1);
    o1 = wmma_f16(p1, v11, o1);
  }

  // residual + relu epilogue, out[b, s, h*64 + d]
  size_t base0 = ((size_t)b * S + rb * 64 + mi * 16 + 8 * hi) * 512 + h * 64 + nb * 16 + n;
  const float* xp = X + base0;
  float* op = out + base0;
#pragma unroll
  for (int r = 0; r < 8; ++r) {
    float r0 = o0[r] + xp[0];
    float r1 = o1[r] + xp[16];
    op[0]  = r0 > 0.0f ? r0 : 0.0f;
    op[16] = r1 > 0.0f ? r1 : 0.0f;
    xp += 512; op += 512;
  }
}

extern "C" void kernel_launch(void* const* d_in, const int* in_sizes, int n_in,
                              void* d_out, int out_size, void* d_ws, size_t ws_size,
                              hipStream_t stream) {
  (void)in_sizes; (void)n_in; (void)out_size; (void)ws_size;
  const float* X   = (const float*)d_in[0];
  const float* Wq  = (const float*)d_in[1];
  const float* Wk  = (const float*)d_in[2];
  const float* Wv  = (const float*)d_in[3];
  const float* gnw = (const float*)d_in[4];
  const float* gnb = (const float*)d_in[5];
  float* out = (float*)d_out;

  const int B = 2, S = 2048, Din = 512, Dm = 512;
  const int M = B * S;

  char* ws = (char*)d_ws;
  size_t off = 0;
  auto alloc = [&](size_t bytes) -> void* {
    void* p = ws + off;
    off = (off + bytes + 255) & ~(size_t)255;
    return p;
  };
  _Float16* Xh  = (_Float16*)alloc((size_t)M * Din * sizeof(_Float16));
  _Float16* WhT = (_Float16*)alloc((size_t)3 * Din * Dm * sizeof(_Float16));
  float*    PF  = (float*)  alloc((size_t)3 * M * Dm * sizeof(float));
  _Float16* qh  = (_Float16*)alloc((size_t)M * Dm * sizeof(_Float16));
  _Float16* kh  = (_Float16*)alloc((size_t)M * Dm * sizeof(_Float16));
  _Float16* vh  = (_Float16*)alloc((size_t)M * Dm * sizeof(_Float16));
  _Float16* vT  = (_Float16*)alloc((size_t)M * Dm * sizeof(_Float16));

  int nX = M * Din;
  f32_to_f16_kernel<<<dim3((nX + 255) / 256), dim3(256), 0, stream>>>(X, Xh, nX);
  convert_transpose_w<<<dim3(8, 8, 3), dim3(256), 0, stream>>>(Wq, Wk, Wv, WhT);

  proj_gemm_f16<<<dim3(Dm / 64, M / 64, 3), dim3(256), 0, stream>>>(Xh, WhT, PF, M, Dm, Din);

  xpos_gn_silu<<<dim3(S, B), dim3(512), 0, stream>>>(
      PF, PF + (size_t)M * Dm, PF + 2 * (size_t)M * Dm, gnw, gnb, qh, kh, vh, S);

  transpose_v<<<dim3(S / 64, B * 8), dim3(256), 0, stream>>>(vh, vT, S);

  attn_msr<<<dim3(S / 64, 8, B), dim3(256), 0, stream>>>(qh, kh, vT, X, out, S);
}
